// SelfAttention_50440095924233
// MI455X (gfx1250) — matrix operand order repaired and checked by reading, not compiled, not run on hardware
//
#include <hip/hip_runtime.h>

// ---------------------------------------------------------------------------
// MI455X (gfx1250) self-attention block.
// Compute-bound (~206 GFLOP vs ~8.6us of HBM traffic at 23.3 TB/s), so all
// matmuls route through v_wmma_f32_16x16x32_bf16 with fp32 accumulation.
// Projections use a bf16 hi/lo split (3 WMMAs / tile) for near-fp32 accuracy.
// GEMM k-loop is software-pipelined (register prefetch of next k-tile hides
// global latency under the WMMAs); flash-attn K tiles use the CDNA5 async
// global->LDS path (ASYNCcnt) since they are raw bf16 copies.
// ---------------------------------------------------------------------------

#define DIMD   3072
#define NHEADS 24
#define HDIM   128
#define SEQ    2048
#define TDIM   9216   // 3*DIM
#define SMS    0.08838834764831845f  // 1/sqrt(128)

typedef unsigned short u16;
typedef __attribute__((ext_vector_type(16))) __bf16 v16bf;
typedef __attribute__((ext_vector_type(8)))  float  v8f;

union FragB { v16bf v; uint4 q[2]; };
union U4    { uint4 u; u16 s[8]; float f[4]; };
union U2    { uint2 u; u16 s[4]; };

__device__ __forceinline__ u16 f2bf(float f) {
    unsigned u = __float_as_uint(f);
    u += 0x7fffu + ((u >> 16) & 1u);          // round-to-nearest-even
    return (u16)(u >> 16);
}
__device__ __forceinline__ float bf2f(u16 h) {
    return __uint_as_float(((unsigned)h) << 16);
}
__device__ __forceinline__ v8f vzero8() {
    v8f z;
#pragma unroll
    for (int i = 0; i < 8; ++i) z[i] = 0.0f;
    return z;
}
__device__ __forceinline__ v8f wmma_bf16(v16bf a, v16bf b, v8f c) {
    return __builtin_amdgcn_wmma_f32_16x16x32_bf16(false, a, false, b,
                                                   (short)0, c, false, false);
}

// ---------------------------------------------------------------------------
// Kernel 1/4: C[M,N] = X[M,K] * W[N,K]^T (+bias), fp32 in/out,
// bf16 hi/lo split WMMA. Block tile 128x128x32, 8 waves of 32x64.
// Pipelined: next k-tile global loads issued while WMMAs run.
// ---------------------------------------------------------------------------
#define BM 128
#define BN 128
#define BK 32
#define KPAD 40   // 80B row stride: conflict-free ds_load_b128 fragments

__global__ __launch_bounds__(256)
void gemm_nt(const float* __restrict__ X, const float* __restrict__ W,
             const float* __restrict__ bias, float* __restrict__ C,
             int N, int K) {
    __shared__ __align__(16) u16 sXh[BM][KPAD];
    __shared__ __align__(16) u16 sXl[BM][KPAD];
    __shared__ __align__(16) u16 sWh[BN][KPAD];
    __shared__ __align__(16) u16 sWl[BN][KPAD];

    const int tid  = threadIdx.x;
    const int lane = tid & 31;
    const int half = lane >> 4;
    const int l16  = lane & 15;
    const int wid  = tid >> 5;
    const int wm   = wid >> 1;   // 0..3  -> rows 32*wm
    const int wn   = wid & 1;    // 0..1  -> cols 64*wn
    const int m0   = blockIdx.y * BM;
    const int n0   = blockIdx.x * BN;

    // this thread's staging slots (4 x float4 in X tile, 4 in W tile)
    int srow[4], scol[4];
#pragma unroll
    for (int i = 0; i < 4; ++i) {
        int f = tid + 256 * i;
        srow[i] = f >> 3;
        scol[i] = (f & 7) * 4;
    }

    U4 xr[4], wr[4];
    auto load_tiles = [&](int k0) {
#pragma unroll
        for (int i = 0; i < 4; ++i) {
            xr[i].u = *(const uint4*)(X + (size_t)(m0 + srow[i]) * K + k0 + scol[i]);
            wr[i].u = *(const uint4*)(W + (size_t)(n0 + srow[i]) * K + k0 + scol[i]);
        }
    };
    auto store_tiles = [&]() {
#pragma unroll
        for (int i = 0; i < 4; ++i) {
            int row = srow[i], c4 = scol[i];
#pragma unroll
            for (int e = 0; e < 4; ++e) {
                float xv = xr[i].f[e], wv = wr[i].f[e];
                u16 xh = f2bf(xv);
                sXh[row][c4 + e] = xh;
                sXl[row][c4 + e] = f2bf(xv - bf2f(xh));
                u16 wh = f2bf(wv);
                sWh[row][c4 + e] = wh;
                sWl[row][c4 + e] = f2bf(wv - bf2f(wh));
            }
        }
    };

    v8f acc[2][4];
#pragma unroll
    for (int mi = 0; mi < 2; ++mi)
#pragma unroll
        for (int ni = 0; ni < 4; ++ni) acc[mi][ni] = vzero8();

    load_tiles(0);                        // prologue

    for (int k0 = 0; k0 < K; k0 += BK) {
        store_tiles();                    // regs -> LDS (bf16 hi/lo)
        __syncthreads();
        if (k0 + BK < K) load_tiles(k0 + BK);   // overlap with WMMAs below

        // ---- fragments (documented 16-bit A(16x32)/B(32x16) layouts) ----
        FragB ah[2], al[2], bh[4], bl[4];
#pragma unroll
        for (int mi = 0; mi < 2; ++mi) {
            int r  = 32 * wm + 16 * mi + l16;
            int cb = half * 8;
            ah[mi].q[0] = *(const uint4*)&sXh[r][cb];
            ah[mi].q[1] = *(const uint4*)&sXh[r][cb + 16];
            al[mi].q[0] = *(const uint4*)&sXl[r][cb];
            al[mi].q[1] = *(const uint4*)&sXl[r][cb + 16];
        }
#pragma unroll
        for (int ni = 0; ni < 4; ++ni) {
            int r  = 64 * wn + 16 * ni + l16;
            int cb = half * 8;
            bh[ni].q[0] = *(const uint4*)&sWh[r][cb];
            bh[ni].q[1] = *(const uint4*)&sWh[r][cb + 16];
            bl[ni].q[0] = *(const uint4*)&sWl[r][cb];
            bl[ni].q[1] = *(const uint4*)&sWl[r][cb + 16];
        }
#pragma unroll
        for (int mi = 0; mi < 2; ++mi)
#pragma unroll
            for (int ni = 0; ni < 4; ++ni) {
                acc[mi][ni] = wmma_bf16(ah[mi].v, bh[ni].v, acc[mi][ni]);
                acc[mi][ni] = wmma_bf16(ah[mi].v, bl[ni].v, acc[mi][ni]);
                acc[mi][ni] = wmma_bf16(al[mi].v, bh[ni].v, acc[mi][ni]);
            }
        __syncthreads();
    }

    // ---- epilogue: f32 C-frag layout (vgpr g -> row g+8*half, col l16) ----
#pragma unroll
    for (int mi = 0; mi < 2; ++mi)
#pragma unroll
        for (int ni = 0; ni < 4; ++ni) {
            int col = n0 + 64 * wn + 16 * ni + l16;
            float b = bias ? bias[col] : 0.0f;
#pragma unroll
            for (int g = 0; g < 8; ++g) {
                int row = m0 + 32 * wm + 16 * mi + g + 8 * half;
                C[(size_t)row * N + col] = acc[mi][ni][g] + b;
            }
        }
}

// ---------------------------------------------------------------------------
// Kernel 2: per-head RMSNorm (q,k) + RoPE + scatter to bf16 [H][L][D].
// q pre-scaled by 1/sqrt(HDIM). One block per sequence position.
// ---------------------------------------------------------------------------
__global__ __launch_bounds__(256)
void qknorm_rope_scatter(const float* __restrict__ qkv, const float* __restrict__ pe,
                         const float* __restrict__ qs, const float* __restrict__ ks,
                         u16* __restrict__ qT, u16* __restrict__ kT, u16* __restrict__ vT) {
    const int l    = blockIdx.x;
    const int lane = threadIdx.x & 31;
    const int wid  = threadIdx.x >> 5;
    const int d0   = lane * 4;

    for (int j = wid; j < 72; j += 8) {           // j = s*24 + h
        int s = j / 24, h = j % 24;
        const float* src = qkv + (size_t)l * TDIM + (size_t)j * HDIM;
        float4 xv = *(const float4*)(src + d0);
        float y[4] = {xv.x, xv.y, xv.z, xv.w};
        u16* dst;
        if (s < 2) {
            float ss = y[0]*y[0] + y[1]*y[1] + y[2]*y[2] + y[3]*y[3];
#pragma unroll
            for (int m = 16; m >= 1; m >>= 1) ss += __shfl_xor(ss, m, 32);
            float rr = rsqrtf(ss * (1.0f / HDIM) + 1e-6f);
            const float* sc = (s == 0) ? qs : ks;
            float yn[4];
#pragma unroll
            for (int e = 0; e < 4; ++e) yn[e] = y[e] * rr * sc[d0 + e];
            // RoPE: out[2p+jj] = pe[p][jj][0]*x[2p] + pe[p][jj][1]*x[2p+1]
            const float* peb = pe + (size_t)l * 256;   // [64][2][2]
#pragma unroll
            for (int e = 0; e < 4; ++e) {
                int p  = lane * 2 + (e >> 1);
                int jj = e & 1;
                const float* pp = peb + (p * 2 + jj) * 2;
                y[e] = pp[0] * yn[e & 2] + pp[1] * yn[(e & 2) + 1];
            }
            if (s == 0) {
#pragma unroll
                for (int e = 0; e < 4; ++e) y[e] *= SMS;
            }
            dst = ((s == 0) ? qT : kT) + ((size_t)h * SEQ + l) * HDIM + d0;
        } else {
            dst = vT + ((size_t)h * SEQ + l) * HDIM + d0;
        }
        U2 o;
#pragma unroll
        for (int e = 0; e < 4; ++e) o.s[e] = f2bf(y[e]);
        *(uint2*)dst = o.u;
    }
}

// ---------------------------------------------------------------------------
// Kernel 3: flash attention. Block = 1 head x 128 q-rows; each wave owns
// 16 q-rows -> online softmax needs no cross-wave reduction. Key tile = 64.
// K tile staged with CDNA5 async global->LDS (ASYNCcnt); V transposed via
// registers (async copies cannot transpose).
// ---------------------------------------------------------------------------
#define TK     64
#define KRPAD  136
#define VTPAD  72
#define PPAD   72

__global__ __launch_bounds__(256)
void flash_attn(const u16* __restrict__ qT, const u16* __restrict__ kT,
                const u16* __restrict__ vT, float* __restrict__ attn) {
    __shared__ __align__(16) u16 sK [TK][KRPAD];     // [key][d], row-major
    __shared__ __align__(16) u16 sVt[HDIM][VTPAD];   // [d][key], transposed
    __shared__ __align__(16) u16 sP [8][16][PPAD];   // per-wave P slab

    const int tid  = threadIdx.x;
    const int lane = tid & 31;
    const int half = lane >> 4;
    const int l16  = lane & 15;
    const int wid  = tid >> 5;
    const int h    = blockIdx.y;
    const int q0   = blockIdx.x * 128 + wid * 16;

    // Q fragments for this wave's 16 rows (K=128 -> 4 frags); pre-scaled.
    FragB qf[4];
    const u16* qb = qT + ((size_t)h * SEQ + q0 + l16) * HDIM;
#pragma unroll
    for (int kk = 0; kk < 4; ++kk) {
        qf[kk].q[0] = *(const uint4*)(qb + kk * 32 + half * 8);
        qf[kk].q[1] = *(const uint4*)(qb + kk * 32 + half * 8 + 16);
    }

    v8f o[8];
#pragma unroll
    for (int nn = 0; nn < 8; ++nn) o[nn] = vzero8();
    float mrun[8], lrun[8];
#pragma unroll
    for (int g = 0; g < 8; ++g) { mrun[g] = -1e30f; lrun[g] = 0.0f; }

    for (int kb = 0; kb < SEQ; kb += TK) {
        __syncthreads();
        // ---- stage K tile via async global->LDS (raw bf16 copy) ----
#pragma unroll
        for (int i = 0; i < 4; ++i) {
            int q = tid + 256 * i;              // 1024 16B slots
            int r = q >> 4, c8 = (q & 15) * 8;
            const u16* gk = kT + ((size_t)h * SEQ + kb + r) * HDIM + c8;
            unsigned lk = (unsigned)(size_t)&sK[r][c8];   // LDS byte offset
            asm volatile("global_load_async_to_lds_b128 %0, %1, off"
                         :: "v"(lk), "v"(gk) : "memory");
            // ---- V tile: register transpose -> sVt[d][key] ----
            U4 t;
            t.u = *(const uint4*)(vT + ((size_t)h * SEQ + kb + r) * HDIM + c8);
#pragma unroll
            for (int e = 0; e < 8; ++e) sVt[c8 + e][r] = t.s[e];
        }
        asm volatile("s_wait_asynccnt 0" ::: "memory");
        __syncthreads();

        // S = Q * K^T  (16 x 64 per wave)
        v8f sfr[4];
#pragma unroll
        for (int ni = 0; ni < 4; ++ni) {
            v8f s = vzero8();
#pragma unroll
            for (int kk = 0; kk < 4; ++kk) {
                FragB bk;
                const u16* kp = &sK[16 * ni + l16][kk * 32 + half * 8];
                bk.q[0] = *(const uint4*)kp;
                bk.q[1] = *(const uint4*)(kp + 16);
                s = wmma_bf16(qf[kk].v, bk.v, s);
            }
            sfr[ni] = s;
        }

        // online softmax (per-row stats live in each 16-lane half)
        float mnew[8], alpha[8], psum[8];
#pragma unroll
        for (int g = 0; g < 8; ++g) {
            float tm = sfr[0][g];
#pragma unroll
            for (int ni = 1; ni < 4; ++ni) tm = fmaxf(tm, sfr[ni][g]);
#pragma unroll
            for (int m = 8; m >= 1; m >>= 1) tm = fmaxf(tm, __shfl_xor(tm, m, 32));
            mnew[g]  = fmaxf(mrun[g], tm);
            alpha[g] = __expf(mrun[g] - mnew[g]);
            psum[g]  = 0.0f;
        }
#pragma unroll
        for (int ni = 0; ni < 4; ++ni)
#pragma unroll
            for (int g = 0; g < 8; ++g) {
                float p = __expf(sfr[ni][g] - mnew[g]);
                psum[g] += p;
                sP[wid][g + 8 * half][ni * 16 + l16] = f2bf(p);
            }
#pragma unroll
        for (int g = 0; g < 8; ++g) {
#pragma unroll
            for (int m = 8; m >= 1; m >>= 1) psum[g] += __shfl_xor(psum[g], m, 32);
            lrun[g] = lrun[g] * alpha[g] + psum[g];
            mrun[g] = mnew[g];
#pragma unroll
            for (int nn = 0; nn < 8; ++nn) o[nn][g] *= alpha[g];
        }

        // O += P * V   (A = P from wave-private slab, B = V^T columns)
#pragma unroll
        for (int k2 = 0; k2 < 2; ++k2) {
            FragB ap;
            const u16* pp = &sP[wid][l16][k2 * 32 + half * 8];
            ap.q[0] = *(const uint4*)pp;
            ap.q[1] = *(const uint4*)(pp + 16);
#pragma unroll
            for (int nn = 0; nn < 8; ++nn) {
                FragB bv;
                const u16* vp = &sVt[nn * 16 + l16][k2 * 32 + half * 8];
                bv.q[0] = *(const uint4*)vp;
                bv.q[1] = *(const uint4*)(vp + 16);
                o[nn] = wmma_bf16(ap.v, bv.v, o[nn]);
            }
        }
    }

    // epilogue: write merged-head fp32 [L][H*D]
#pragma unroll
    for (int nn = 0; nn < 8; ++nn)
#pragma unroll
        for (int g = 0; g < 8; ++g) {
            int row = q0 + g + 8 * half;
            attn[(size_t)row * DIMD + h * HDIM + nn * 16 + l16] =
                o[nn][g] / lrun[g];
        }
}

// ---------------------------------------------------------------------------
// Launch. Workspace: qkv fp32 (75.5MB, reused for attention output) +
// q/k/v bf16 [H][L][D] (37.7MB) = ~113MB.
// ---------------------------------------------------------------------------
extern "C" void kernel_launch(void* const* d_in, const int* in_sizes, int n_in,
                              void* d_out, int out_size, void* d_ws, size_t ws_size,
                              hipStream_t stream) {
    (void)in_sizes; (void)n_in; (void)out_size; (void)ws_size;
    const float* x       = (const float*)d_in[0];
    const float* pe      = (const float*)d_in[1];
    const float* w_qkv   = (const float*)d_in[2];
    const float* w_proj  = (const float*)d_in[3];
    const float* b_proj  = (const float*)d_in[4];
    const float* q_scale = (const float*)d_in[5];
    const float* k_scale = (const float*)d_in[6];
    float* out = (float*)d_out;

    char*  ws        = (char*)d_ws;
    float* qkv       = (float*)ws;                       // [SEQ][TDIM]
    size_t qkv_bytes = (size_t)SEQ * TDIM * sizeof(float);
    u16*   qT = (u16*)(ws + qkv_bytes);                  // [H][L][D] bf16
    u16*   kT = qT + (size_t)NHEADS * SEQ * HDIM;
    u16*   vT = kT + (size_t)NHEADS * SEQ * HDIM;
    float* attn = qkv;                                   // reuse qkv region

    gemm_nt<<<dim3(TDIM / BN, SEQ / BM), 256, 0, stream>>>(
        x, w_qkv, nullptr, qkv, TDIM, DIMD);
    qknorm_rope_scatter<<<dim3(SEQ), 256, 0, stream>>>(
        qkv, pe, q_scale, k_scale, qT, kT, vT);
    flash_attn<<<dim3(SEQ / 128, NHEADS), 256, 0, stream>>>(
        qT, kT, vT, attn);
    gemm_nt<<<dim3(DIMD / BN, SEQ / BM), 256, 0, stream>>>(
        attn, w_proj, b_proj, out, DIMD, DIMD);
}
